// Transformer_12584254177825
// MI455X (gfx1250) — matrix operand order repaired and checked by reading, not compiled, not run on hardware
//
#include <hip/hip_runtime.h>
#include <hip/hip_bf16.h>

// ---------------------------------------------------------------------------
// CDNA5 (gfx1250) transformer forward.  All GEMMs run on v_wmma_f32_16x16x32_bf16
// (f32 accumulate).  Tiles are staged f32->bf16 into double-buffered LDS in
// fragment-major layout: staging is batched global_load_b128 (12 in flight) +
// v_cvt_pk_bf16_f32 + ds_store_b64; fragment gathering is pure ds_load_b128.
// ---------------------------------------------------------------------------

typedef __bf16 bf16_t;
typedef __attribute__((ext_vector_type(16))) __bf16 v16bf;
typedef __attribute__((ext_vector_type(8)))  __bf16 v8bf;
typedef __attribute__((ext_vector_type(4)))  __bf16 v4bf;
typedef __attribute__((ext_vector_type(8)))  float  v8f;

#define D_MODEL 512
#define NHEAD   8
#define DKDIM   64
#define DFF     2048
#define NLAYER  6
#define VTOUT   10000
#define BSZ     32
#define SEQ     256

// tile geometry: block = 128 threads (4 wave32), block tile 128(M) x 64(N),
// K-step 32.  Each wave owns a 32x64 strip = 2 x 4 WMMA accumulators.
#define BM 128
#define BN 64
#define BK 32
#define PITCH 40   // elements per LDS row (32 K + pad); 80 bytes, 16B aligned

// ---------------- embedding + positional encoding --------------------------
__global__ __launch_bounds__(256) void embed_pe_kernel(
    const int* __restrict__ tok, const float* __restrict__ emb,
    float* __restrict__ out, int seqlen)
{
  int row = blockIdx.x;            // b*seqlen + s
  int s   = row % seqlen;
  int tkn = tok[row];
  const float ln10k = 9.210340371976184f;       // ln(10000)
  const float scale = 22.62741699796952f;       // sqrt(512)
  for (int d = threadIdx.x; d < D_MODEL; d += 256) {
    float ang = (float)s * expf(-(2.0f * (float)d / (float)D_MODEL) * ln10k);
    float pe  = (d & 1) ? cosf(ang) : sinf(ang);
    out[(long)row * D_MODEL + d] = emb[(long)tkn * D_MODEL + d] * scale + pe;
  }
}

// ---------------- layernorm (row length 512, block 256) ---------------------
__global__ __launch_bounds__(256) void layernorm_kernel(
    const float* __restrict__ x, const float* __restrict__ g,
    const float* __restrict__ b, float* __restrict__ y)
{
  __shared__ float red[256];
  int row = blockIdx.x, t = threadIdx.x;
  const float* xr = x + (long)row * D_MODEL;
  float a0 = xr[t], a1 = xr[t + 256];
  red[t] = a0 + a1; __syncthreads();
  for (int o = 128; o; o >>= 1) { if (t < o) red[t] += red[t + o]; __syncthreads(); }
  float mean = red[0] * (1.0f / D_MODEL); __syncthreads();
  red[t] = a0 * a0 + a1 * a1; __syncthreads();
  for (int o = 128; o; o >>= 1) { if (t < o) red[t] += red[t + o]; __syncthreads(); }
  float var  = red[0] * (1.0f / D_MODEL) - mean * mean;
  float rstd = rsqrtf(var + 1e-6f);
  float* yr = y + (long)row * D_MODEL;
  yr[t]       = (a0 - mean) * rstd * g[t]       + b[t];
  yr[t + 256] = (a1 - mean) * rstd * g[t + 256] + b[t + 256];
}

// ---------------- masked softmax over k (k-len == 256 always) ---------------
__global__ __launch_bounds__(256) void masked_softmax_kernel(
    float* __restrict__ s, const int* __restrict__ mask,
    long mb_stride, long mq_stride, int nh, float scale)
{
  __shared__ float red[256];
  int q = blockIdx.x, z = blockIdx.y, t = threadIdx.x;
  int b = z / nh;
  float* row = s + ((long)z * gridDim.x + q) * 256;
  const int* mrow = mask + (long)b * mb_stride + (long)q * mq_stride;
  float v = row[t] * scale;
  if (mrow[t] == 0) v = -1.0e9f;
  red[t] = v; __syncthreads();
  for (int o = 128; o; o >>= 1) { if (t < o) red[t] = fmaxf(red[t], red[t + o]); __syncthreads(); }
  float mx = red[0]; __syncthreads();
  float e = expf(v - mx);
  red[t] = e; __syncthreads();
  for (int o = 128; o; o >>= 1) { if (t < o) red[t] += red[t + o]; __syncthreads(); }
  row[t] = e / red[0];
}

// ---------------- row log-softmax (in place) --------------------------------
__global__ __launch_bounds__(256) void log_softmax_kernel(
    float* __restrict__ x, int ncol)
{
  __shared__ float red[256];
  int row = blockIdx.x, t = threadIdx.x;
  float* xr = x + (long)row * ncol;
  float mx = -3.4e38f;
  for (int c = t; c < ncol; c += 256) mx = fmaxf(mx, xr[c]);
  red[t] = mx; __syncthreads();
  for (int o = 128; o; o >>= 1) { if (t < o) red[t] = fmaxf(red[t], red[t + o]); __syncthreads(); }
  mx = red[0]; __syncthreads();
  float sum = 0.0f;
  for (int c = t; c < ncol; c += 256) sum += expf(xr[c] - mx);
  red[t] = sum; __syncthreads();
  for (int o = 128; o; o >>= 1) { if (t < o) red[t] += red[t + o]; __syncthreads(); }
  float lse = mx + logf(red[0]);
  for (int c = t; c < ncol; c += 256) xr[c] -= lse;
}

// ---------------- batched bf16 WMMA GEMM ------------------------------------
// C[m,n] = sum_k A[m,k] * (bTrans ? B[n,k] : B[k,n])   + bias/resid/relu
// Dual batch stride: z = blockIdx.z -> (bb = z/nh, hh = z%nh).
// M must be a multiple of 128; N edge-guarded; K must be a multiple of 32.
// Double-buffered LDS: stage tile i+1 while WMMAs consume tile i.
__global__ __launch_bounds__(128) void gemm_bf16_wmma_kernel(
    const float* __restrict__ A, const float* __restrict__ B,
    float* __restrict__ C, const float* __restrict__ bias,
    const float* __restrict__ resid,
    int M, int N, int K, int lda, int ldb, int ldc,
    long sAb, long sAh, long sBb, long sBh, long sCb, long sCh,
    int nh, int bTrans, int relu)
{
  __shared__ bf16_t As[2][BM * PITCH];   // [m][k], 2 x 10240 B
  __shared__ bf16_t Bs[2][BN * PITCH];   // [n][k] always, 2 x 5120 B

  const int z = blockIdx.z, bb = z / nh, hh = z % nh;
  const float* Ab = A + (long)bb * sAb + (long)hh * sAh;
  const float* Bb = B + (long)bb * sBb + (long)hh * sBh;
  float*       Cb = C + (long)bb * sCb + (long)hh * sCh;
  const float* Rb = resid ? resid + (long)bb * sCb + (long)hh * sCh : nullptr;

  const int n0 = blockIdx.x * BN, m0 = blockIdx.y * BM;
  const int tid = threadIdx.x, lane = tid & 31, wave = tid >> 5;

  v8f acc[2][4];
#pragma unroll
  for (int mi = 0; mi < 2; ++mi)
#pragma unroll
    for (int t = 0; t < 4; ++t)
#pragma unroll
      for (int r = 0; r < 8; ++r) acc[mi][t][r] = 0.0f;

  const int kbA  = (lane < 16) ? 0 : 8;    // A fragment K-half base
  const int kbB  = (lane < 16) ? 0 : 8;    // B fragment K-half base
  const int lidx = lane & 15;

  // stage one 128x32 A tile + 64x32 B tile (f32 -> bf16) into LDS buffer `buf`.
  // Phase 1 issues all 12 global_load_b128 (kept in flight together);
  // phase 2 converts + ds_store_b64 after a single loadcnt wait.
  auto stage = [&](int buf, int kk) {
    float4 va[8];
#pragma unroll
    for (int j = 0; j < 8; ++j) {
      int f = tid + 128 * j;
      int r = f >> 3, kq = (f & 7) * 4;
      const float* ap = Ab + (long)(m0 + r) * lda + kk + kq;
      va[j] = *reinterpret_cast<const float4*>(ap);
      __builtin_prefetch(ap + BK, 0, 1);   // speculative, no guard needed
    }
    float4 vb[4];
    if (bTrans) {            // memory [N,K]: straight row copy
#pragma unroll
      for (int j = 0; j < 4; ++j) {
        int f = tid + 128 * j;
        int n = f >> 3, kq = (f & 7) * 4;
        int gn = n0 + n;
        vb[j] = make_float4(0.f, 0.f, 0.f, 0.f);
        if (gn < N) {
          const float* bp = Bb + (long)gn * ldb + kk + kq;
          vb[j] = *reinterpret_cast<const float4*>(bp);
          __builtin_prefetch(bp + BK, 0, 1);
        }
      }
    } else {                 // memory [K,N]: transpose on the way in
#pragma unroll
      for (int j = 0; j < 4; ++j) {
        int f = tid + 128 * j;
        int n = f >> 3, kq = (f & 7) * 4;
        int gn = n0 + n;
        vb[j] = make_float4(0.f, 0.f, 0.f, 0.f);
        if (gn < N) {
          const float* bp = Bb + (long)(kk + kq) * ldb + gn;
          vb[j].x = bp[0];
          vb[j].y = bp[(long)ldb];
          vb[j].z = bp[(long)ldb * 2];
          vb[j].w = bp[(long)ldb * 3];
        }
      }
    }
    // phase 2: convert + store (single wait for all loads above)
#pragma unroll
    for (int j = 0; j < 8; ++j) {
      int f = tid + 128 * j;
      int r = f >> 3, kq = (f & 7) * 4;
      v4bf pk;
      pk[0] = (bf16_t)va[j].x; pk[1] = (bf16_t)va[j].y;
      pk[2] = (bf16_t)va[j].z; pk[3] = (bf16_t)va[j].w;
      *reinterpret_cast<v4bf*>(&As[buf][r * PITCH + kq]) = pk;
    }
#pragma unroll
    for (int j = 0; j < 4; ++j) {
      int f = tid + 128 * j;
      int n = f >> 3, kq = (f & 7) * 4;
      v4bf pk;
      pk[0] = (bf16_t)vb[j].x; pk[1] = (bf16_t)vb[j].y;
      pk[2] = (bf16_t)vb[j].z; pk[3] = (bf16_t)vb[j].w;
      *reinterpret_cast<v4bf*>(&Bs[buf][n * PITCH + kq]) = pk;
    }
  };

  union U16 { v16bf v; v8bf h[2]; };

  const int ntiles = K / BK;
  stage(0, 0);
  for (int i = 0; i < ntiles; ++i) {
    __syncthreads();                       // tile i fully staged & visible
    if (i + 1 < ntiles) stage((i + 1) & 1, (i + 1) * BK);  // overlap next tile

    const int buf = i & 1;
    // ---- fragments: 12 ds_load_b128 issued together, then 8 WMMAs ----
    // A 16x32 bf16 (ISA layout): lanes 0-15 K 0..7 / 16..23,
    // lanes 16-31 K 8..15 / 24..31 -> two contiguous 8-elem runs.
    U16 afrag[2];
#pragma unroll
    for (int mi = 0; mi < 2; ++mi) {
      int mrow = wave * 32 + mi * 16 + lidx;
      const bf16_t* arow = &As[buf][mrow * PITCH + kbA];
      afrag[mi].h[0] = *reinterpret_cast<const v8bf*>(arow);
      afrag[mi].h[1] = *reinterpret_cast<const v8bf*>(arow + 16);
    }
    // B 32x16 bf16: N = lane%16, lanes 0-15 K 0..7 / 16..23,
    // lanes 16-31 K 8..15 / 24..31 -> two contiguous 8-elem runs.
    U16 bfrag[4];
#pragma unroll
    for (int t = 0; t < 4; ++t) {
      const bf16_t* brow = &Bs[buf][(t * 16 + lidx) * PITCH + kbB];
      bfrag[t].h[0] = *reinterpret_cast<const v8bf*>(brow);
      bfrag[t].h[1] = *reinterpret_cast<const v8bf*>(brow + 16);
    }
#pragma unroll
    for (int t = 0; t < 4; ++t)
#pragma unroll
      for (int mi = 0; mi < 2; ++mi)
        acc[mi][t] = __builtin_amdgcn_wmma_f32_16x16x32_bf16(
            false, afrag[mi].v, false, bfrag[t].v, (short)0, acc[mi][t],
            false, false);
  }

  // ---- epilogue: bias + residual + relu, N edge-guarded ----
#pragma unroll
  for (int t = 0; t < 4; ++t) {
    int gn = n0 + t * 16 + lidx;
    if (gn >= N) continue;
    float bv = bias ? bias[gn] : 0.0f;
#pragma unroll
    for (int mi = 0; mi < 2; ++mi) {
      int gm = m0 + wave * 32 + mi * 16 + ((lane < 16) ? 0 : 8);
#pragma unroll
      for (int r = 0; r < 8; ++r) {
        long ci = (long)(gm + r) * ldc + gn;
        float v = acc[mi][t][r] + bv;
        if (Rb) v += Rb[ci];
        if (relu) v = fmaxf(v, 0.0f);
        Cb[ci] = v;
      }
    }
  }
}

// ---------------------------------------------------------------------------
// Host-side orchestration
// ---------------------------------------------------------------------------
struct MHAp { const float *wq, *bq, *wk, *bk, *wv, *bv, *w0, *b0; };

static inline void launch_gemm(hipStream_t st,
    const float* A, const float* B, float* C,
    const float* bias, const float* resid,
    int M, int N, int K, int lda, int ldb, int ldc,
    int batch, int nh,
    long sAb, long sAh, long sBb, long sBh, long sCb, long sCh,
    int bTrans, int relu)
{
  dim3 grid((N + BN - 1) / BN, M / BM, batch);
  gemm_bf16_wmma_kernel<<<grid, 128, 0, st>>>(A, B, C, bias, resid,
      M, N, K, lda, ldb, ldc, sAb, sAh, sBb, sBh, sCb, sCh, nh, bTrans, relu);
}

static void run_mha(hipStream_t st, const float* qin, const float* kvin,
                    int qlen, int klen, const MHAp& p,
                    const int* mask, long mb, long mq,
                    const float* resid, float* xout,
                    float* Qb, float* Kb, float* Vb, float* big,
                    float* aw_copy)
{
  const int Mq = BSZ * qlen, Mk = BSZ * klen;
  // projections
  launch_gemm(st, qin,  p.wq, Qb, p.bq, nullptr, Mq, D_MODEL, D_MODEL,
              D_MODEL, D_MODEL, D_MODEL, 1, 1, 0,0,0,0,0,0, 1, 0);
  launch_gemm(st, kvin, p.wk, Kb, p.bk, nullptr, Mk, D_MODEL, D_MODEL,
              D_MODEL, D_MODEL, D_MODEL, 1, 1, 0,0,0,0,0,0, 1, 0);
  launch_gemm(st, kvin, p.wv, Vb, p.bv, nullptr, Mk, D_MODEL, D_MODEL,
              D_MODEL, D_MODEL, D_MODEL, 1, 1, 0,0,0,0,0,0, 1, 0);
  // scores[b,h,q,k] = Q . K  (B treated as [N,K] -> transposed mode)
  launch_gemm(st, Qb, Kb, big, nullptr, nullptr, qlen, klen, DKDIM,
              D_MODEL, D_MODEL, klen,
              BSZ * NHEAD, NHEAD,
              (long)qlen * D_MODEL, DKDIM,
              (long)klen * D_MODEL, DKDIM,
              (long)NHEAD * qlen * klen, (long)qlen * klen,
              1, 0);
  masked_softmax_kernel<<<dim3(qlen, BSZ * NHEAD), 256, 0, st>>>(
      big, mask, mb, mq, NHEAD, 0.125f);
  if (aw_copy)
    hipMemcpyAsync(aw_copy, big,
                   (size_t)BSZ * NHEAD * qlen * klen * sizeof(float),
                   hipMemcpyDeviceToDevice, st);
  // context = probs @ V  (normal mode), heads concat into [b,q,D]; reuse Qb
  launch_gemm(st, big, Vb, Qb, nullptr, nullptr, qlen, DKDIM, klen,
              klen, D_MODEL, D_MODEL,
              BSZ * NHEAD, NHEAD,
              (long)NHEAD * qlen * klen, (long)qlen * klen,
              (long)klen * D_MODEL, DKDIM,
              (long)qlen * D_MODEL, DKDIM,
              0, 0);
  // output projection + residual
  launch_gemm(st, Qb, p.w0, xout, p.b0, resid, Mq, D_MODEL, D_MODEL,
              D_MODEL, D_MODEL, D_MODEL, 1, 1, 0,0,0,0,0,0, 1, 0);
}

static inline MHAp make_mha(const float* const* L, int o)
{
  // leaves sorted: b0,bk,bq,bv,w0,wk,wq,wv
  MHAp m;
  m.b0 = L[o + 0]; m.bk = L[o + 1]; m.bq = L[o + 2]; m.bv = L[o + 3];
  m.w0 = L[o + 4]; m.wk = L[o + 5]; m.wq = L[o + 6]; m.wv = L[o + 7];
  return m;
}

extern "C" void kernel_launch(void* const* d_in, const int* in_sizes, int n_in,
                              void* d_out, int out_size, void* d_ws, size_t ws_size,
                              hipStream_t stream)
{
  (void)in_sizes; (void)n_in; (void)out_size; (void)ws_size;

  const int* src_input = (const int*)d_in[0];
  const int* trg_input = (const int*)d_in[1];
  const int* e_mask    = (const int*)d_in[2];   // [B,1,S]
  const int* d_mask    = (const int*)d_in[3];   // [B,T,T]

  // params flattened as a pytree: dicts sorted alphabetically at every level.
  const float* L[260];
  for (int i = 0; i < 260; ++i) L[i] = (const float*)d_in[4 + i];

  // decoder layers: leaves per layer (26): ca(8), ff(b1,b2,w1,w2), ln1b,ln1g,
  // ln2b,ln2g,ln3b,ln3g, sa(8)
  struct DecL { MHAp sa, ca; const float *w1,*b1,*w2,*b2;
                const float *ln1g,*ln1b,*ln2g,*ln2b,*ln3g,*ln3b; } dec[NLAYER];
  for (int i = 0; i < NLAYER; ++i) {
    int o = i * 26;
    dec[i].ca   = make_mha(L, o + 0);
    dec[i].b1   = L[o + 8];  dec[i].b2 = L[o + 9];
    dec[i].w1   = L[o + 10]; dec[i].w2 = L[o + 11];
    dec[i].ln1b = L[o + 12]; dec[i].ln1g = L[o + 13];
    dec[i].ln2b = L[o + 14]; dec[i].ln2g = L[o + 15];
    dec[i].ln3b = L[o + 16]; dec[i].ln3g = L[o + 17];
    dec[i].sa   = make_mha(L, o + 18);
  }
  const float* dec_lnb = L[156];
  const float* dec_lng = L[157];
  // encoder layers: leaves per layer (16): ff(4), ln1b,ln1g,ln2b,ln2g, sa(8)
  struct EncL { MHAp sa; const float *w1,*b1,*w2,*b2;
                const float *ln1g,*ln1b,*ln2g,*ln2b; } enc[NLAYER];
  for (int i = 0; i < NLAYER; ++i) {
    int o = 158 + i * 16;
    enc[i].b1   = L[o + 0]; enc[i].b2 = L[o + 1];
    enc[i].w1   = L[o + 2]; enc[i].w2 = L[o + 3];
    enc[i].ln1b = L[o + 4]; enc[i].ln1g = L[o + 5];
    enc[i].ln2b = L[o + 6]; enc[i].ln2g = L[o + 7];
    enc[i].sa   = make_mha(L, o + 8);
  }
  const float* enc_lnb = L[254];
  const float* enc_lng = L[255];
  const float* out_b   = L[256];
  const float* out_w   = L[257];
  const float* src_emb = L[258];
  const float* trg_emb = L[259];

  // ---- workspace carve-up (fp32) ----
  const long BSD  = (long)BSZ * SEQ * D_MODEL;        // 4,194,304
  float* X   = (float*)d_ws;
  float* X2  = X   + BSD;
  float* T1  = X2  + BSD;
  float* Qb  = T1  + BSD;
  float* Kb  = Qb  + BSD;
  float* Vb  = Kb  + BSD;
  float* EO  = Vb  + BSD;
  float* BIG = EO  + BSD;  // 16,777,216 floats: scores [B,H,S,S] / FF [B*S,DFF]

  const int ROWS = BSZ * SEQ;   // 8192

  // ================= encoder =================
  embed_pe_kernel<<<ROWS, 256, 0, stream>>>(src_input, src_emb, X, SEQ);
  float* cur = X;
  float* alt = X2;
  for (int i = 0; i < NLAYER; ++i) {
    layernorm_kernel<<<ROWS, 256, 0, stream>>>(cur, enc[i].ln1g, enc[i].ln1b, T1);
    run_mha(stream, T1, T1, SEQ, SEQ, enc[i].sa, e_mask, (long)SEQ, 0L,
            cur, alt, Qb, Kb, Vb, BIG, nullptr);                  // x -> alt
    layernorm_kernel<<<ROWS, 256, 0, stream>>>(alt, enc[i].ln2g, enc[i].ln2b, T1);
    launch_gemm(stream, T1, enc[i].w1, BIG, enc[i].b1, nullptr,
                ROWS, DFF, D_MODEL, D_MODEL, D_MODEL, DFF,
                1, 1, 0,0,0,0,0,0, 1, 1);                          // relu
    launch_gemm(stream, BIG, enc[i].w2, cur, enc[i].b2, alt,
                ROWS, D_MODEL, DFF, DFF, DFF, D_MODEL,
                1, 1, 0,0,0,0,0,0, 1, 0);                          // x -> cur
  }
  layernorm_kernel<<<ROWS, 256, 0, stream>>>(cur, enc_lng, enc_lnb, EO);

  // ================= decoder =================
  embed_pe_kernel<<<ROWS, 256, 0, stream>>>(trg_input, trg_emb, X, SEQ);
  cur = X; alt = X2;
  float* aw_dst = (float*)d_out + (long)BSZ * SEQ * VTOUT;
  for (int i = 0; i < NLAYER; ++i) {
    // self attention (causal mask)
    layernorm_kernel<<<ROWS, 256, 0, stream>>>(cur, dec[i].ln1g, dec[i].ln1b, T1);
    run_mha(stream, T1, T1, SEQ, SEQ, dec[i].sa,
            d_mask, (long)SEQ * SEQ, (long)SEQ,
            cur, alt, Qb, Kb, Vb, BIG, nullptr);                   // x -> alt
    // cross attention (encoder mask); last layer's probs are the aw output
    layernorm_kernel<<<ROWS, 256, 0, stream>>>(alt, dec[i].ln2g, dec[i].ln2b, T1);
    run_mha(stream, T1, EO, SEQ, SEQ, dec[i].ca,
            e_mask, (long)SEQ, 0L,
            alt, cur, Qb, Kb, Vb, BIG,
            (i == NLAYER - 1) ? aw_dst : nullptr);                 // x -> cur
    // feed-forward
    layernorm_kernel<<<ROWS, 256, 0, stream>>>(cur, dec[i].ln3g, dec[i].ln3b, T1);
    launch_gemm(stream, T1, dec[i].w1, BIG, dec[i].b1, nullptr,
                ROWS, DFF, D_MODEL, D_MODEL, D_MODEL, DFF,
                1, 1, 0,0,0,0,0,0, 1, 1);
    launch_gemm(stream, BIG, dec[i].w2, alt, dec[i].b2, cur,
                ROWS, D_MODEL, DFF, DFF, DFF, D_MODEL,
                1, 1, 0,0,0,0,0,0, 1, 0);                          // x -> alt
    float* tmp = cur; cur = alt; alt = tmp;                        // x back in cur
  }
  layernorm_kernel<<<ROWS, 256, 0, stream>>>(cur, dec_lng, dec_lnb, T1);

  // logits + log-softmax directly in d_out
  launch_gemm(stream, T1, out_w, (float*)d_out, out_b, nullptr,
              ROWS, VTOUT, D_MODEL, D_MODEL, D_MODEL, VTOUT,
              1, 1, 0,0,0,0,0,0, 1, 0);
  log_softmax_kernel<<<ROWS, 256, 0, stream>>>((float*)d_out, VTOUT);
}
